// RNNModel_57071525429824
// MI455X (gfx1250) — matrix operand order repaired and checked by reading, not compiled, not run on hardware
//
#include <hip/hip_runtime.h>
#include <cmath>

#define M_    64
#define N_    2048
#define T_    512
#define NIN_  2
#define NOUT_ 2
#define TP1_  513
#define ALPHA_ 0.2f          // DT/TAU = 0.02/0.1
#define NWG_  16
#define KC_   256            // K-chunk staged in LDS
#define NKC_  (N_ / KC_)     // 8
#define LDSA_STRIDE 264      // 256 + 8 pad (132 dwords -> bank shift 4, conflict free)
#define OTILE_STRIDE 136     // 128 + 8 pad

typedef __attribute__((ext_vector_type(16))) __bf16 v16bf;
typedef __attribute__((ext_vector_type(8)))  float  v8f;
typedef __attribute__((ext_vector_type(4)))  unsigned int u32x4;

union FragAB { v16bf v; u32x4 q[2]; };

__device__ __forceinline__ unsigned short f2bf(float f) {
  union { float f; unsigned u; } c; c.f = f;
  unsigned r = c.u + 0x7FFFu + ((c.u >> 16) & 1u);   // round-to-nearest-even
  return (unsigned short)(r >> 16);
}

// ---------------------------------------------------------------------------
// Init: convert W_rec to bf16, init u double-buffer, write t=0 planes, reset
// the grid barrier (must happen every launch: graph replays don't re-poison ws)
// ---------------------------------------------------------------------------
__global__ __launch_bounds__(256) void init_kernel(
    const float* __restrict__ Wrec, const float* __restrict__ x0,
    unsigned short* __restrict__ wrec_bf, unsigned short* __restrict__ ubuf0,
    float* __restrict__ out, unsigned* __restrict__ bar) {
  size_t idx = (size_t)blockIdx.x * 256 + threadIdx.x;
  if (idx < (size_t)N_ * N_) wrec_bf[idx] = f2bf(Wrec[idx]);
  if (idx < (size_t)M_ * N_) {
    float xi = x0[0];
    float u0 = tanhf(xi);
    ubuf0[idx] = f2bf(u0);
    size_t m = idx / N_, n = idx % N_;
    size_t o = m * (size_t)N_ * TP1_ + n * (size_t)TP1_;   // t = 0 plane
    out[o] = xi;
    out[(size_t)M_ * N_ * TP1_ + o] = u0;
  }
  if (idx == 0) { bar[0] = 0; bar[1] = 0; }
}

// ---------------------------------------------------------------------------
// Persistent RNN kernel: 16 WGs x 256 threads (8 wave32s). Each wave owns a
// 64x16 tile of the hidden state, kept in registers as 4 WMMA f32 accumulators.
// Per step: D = U(64x2048) x Wrec_bf^T(2048xcols) via v_wmma_f32_16x16x32_bf16,
// A staged through LDS, then Euler update + tanh, coalesced u_new write via an
// LDS repack tile, then a global two-phase barrier.
// ---------------------------------------------------------------------------
__global__ __launch_bounds__(256) void rnn_step_kernel(
    const float* __restrict__ Iin, const float* __restrict__ x0,
    const float* __restrict__ W_in, const float* __restrict__ b_rec,
    const unsigned short* __restrict__ wrec_bf,
    unsigned short* __restrict__ ubufA, unsigned short* __restrict__ ubufB,
    float* __restrict__ xout, float* __restrict__ uout,
    unsigned long long mstr, unsigned long long nstr, unsigned long long tstr,
    unsigned* __restrict__ bar) {
  __shared__ __align__(16) unsigned short ldsA[M_ * LDSA_STRIDE];   // 33792 B
  __shared__ __align__(16) unsigned short otile[M_ * OTILE_STRIDE]; // 17408 B
  __shared__ float Ibuf[M_ * NIN_];                                 //   512 B

  const int tid   = threadIdx.x;
  const int wg    = blockIdx.x;
  const int wave  = tid >> 5;
  const int lane  = tid & 31;
  const int l16   = lane & 15;
  const int lhalf = lane >> 4;
  const int n_loc = wave * 16 + l16;      // 0..127 within WG
  const int n_col = wg * 128 + n_loc;     // global hidden index (per-lane const)

  const float win0  = W_in[n_col * 2 + 0];
  const float win1  = W_in[n_col * 2 + 1];
  const float brec  = b_rec[n_col];
  const float xinit = x0[0];

  // x state: xs[mt][r] <-> row m = mt*16 + lhalf*8 + r, col n_col  (WMMA C/D layout)
  v8f xs[4];
#pragma unroll
  for (int mt = 0; mt < 4; ++mt)
#pragma unroll
    for (int r = 0; r < 8; ++r) xs[mt][r] = xinit;

  unsigned* gen = bar + 1;

#pragma unroll 1
  for (int t = 0; t < T_; ++t) {
    const unsigned short* u_rd = (t & 1) ? ubufB : ubufA;
    unsigned short* u_wr       = (t & 1) ? ubufA : ubufB;

    v8f acc[4];
#pragma unroll
    for (int mt = 0; mt < 4; ++mt)
#pragma unroll
      for (int r = 0; r < 8; ++r) acc[mt][r] = 0.0f;

    // NOTE: kept rolled (#pragma unroll 1): full unroll makes all W_rec
    // fragment addresses compile-time constants, LICM then hoists 2 KB/lane of
    // loop-invariant B fragments out of the t-loop and spills them to scratch.
#pragma unroll 1
    for (int kc = 0; kc < NKC_; ++kc) {
      __syncthreads();                       // prior chunk reads complete
      // cooperative load: 64 rows x 256 bf16 of u into LDS (coalesced b128)
#pragma unroll
      for (int i = 0; i < 8; ++i) {
        int c = tid + i * 256;               // chunk id, 8 bf16 per chunk
        int mrow = c >> 5;
        int k8 = (c & 31) << 3;
        u32x4 d = *(const u32x4*)(u_rd + (size_t)mrow * N_ + kc * KC_ + k8);
        *(u32x4*)(&ldsA[mrow * LDSA_STRIDE + k8]) = d;
      }
      if (kc == 0 && tid < M_ * NIN_) Ibuf[tid] = Iin[(size_t)tid * T_ + t];
      __syncthreads();

#pragma unroll
      for (int ki = 0; ki < KC_ / 32; ++ki) {
        const int kb = ki * 32;
        // B fragment: col n = l16, 16 contiguous K values from Wrec row n_col
        FragAB bfrag;
        const unsigned short* bp =
            wrec_bf + (size_t)n_col * N_ + kc * KC_ + kb + lhalf * 16;
        bfrag.q[0] = *(const u32x4*)(bp);
        bfrag.q[1] = *(const u32x4*)(bp + 8);
        // Load ALL four A fragments into distinct registers first so the
        // four WMMAs can issue back-to-back (disjoint accumulators, shared
        // sources -> no D->A/B hazard, single dscnt wait instead of four).
        FragAB afrag[4];
#pragma unroll
        for (int mt = 0; mt < 4; ++mt) {
          const unsigned short* ap =
              &ldsA[(mt * 16 + l16) * LDSA_STRIDE + kb + lhalf * 16];
          afrag[mt].q[0] = *(const u32x4*)(ap);
          afrag[mt].q[1] = *(const u32x4*)(ap + 8);
        }
#pragma unroll
        for (int mt = 0; mt < 4; ++mt) {
          acc[mt] = __builtin_amdgcn_wmma_f32_16x16x32_bf16(
              false, afrag[mt].v, false, bfrag.v, (short)0, acc[mt], false, false);
        }
      }
    }

    // Euler update + tanh + outputs
#pragma unroll
    for (int mt = 0; mt < 4; ++mt) {
#pragma unroll
      for (int r = 0; r < 8; ++r) {
        int m = mt * 16 + lhalf * 8 + r;
        float ext = Ibuf[2 * m] * win0 + Ibuf[2 * m + 1] * win1 + brec;
        float xv = xs[mt][r];
        float xn = xv + ALPHA_ * (acc[mt][r] + ext - xv);
        float un = tanhf(xn);
        xs[mt][r] = xn;
        size_t off = (size_t)m * mstr + (size_t)n_col * nstr + (size_t)t * tstr;
        xout[off] = xn;
        uout[off] = un;
        otile[m * OTILE_STRIDE + n_loc] = f2bf(un);
      }
    }
    __syncthreads();
    // coalesced write of this WG's 64x128 bf16 u_new tile to the ping-pong buffer
#pragma unroll
    for (int i = 0; i < 4; ++i) {
      int c = tid + i * 256;
      int mrow = c >> 4;
      int n8 = (c & 15) << 3;
      *(u32x4*)(u_wr + (size_t)mrow * N_ + wg * 128 + n8) =
          *(const u32x4*)(&otile[mrow * OTILE_STRIDE + n8]);
    }

    // -------- grid barrier (two-phase: counter + generation) --------
    __threadfence();
    __syncthreads();
    if (tid == 0) {
      unsigned target = (unsigned)(t + 1);
      unsigned arr = __hip_atomic_fetch_add(bar, 1u, __ATOMIC_ACQ_REL,
                                            __HIP_MEMORY_SCOPE_AGENT);
      if (arr == NWG_ - 1) {
        __hip_atomic_store(bar, 0u, __ATOMIC_RELAXED, __HIP_MEMORY_SCOPE_AGENT);
        __hip_atomic_store(gen, target, __ATOMIC_RELEASE, __HIP_MEMORY_SCOPE_AGENT);
      } else {
        while (__hip_atomic_load(gen, __ATOMIC_ACQUIRE,
                                 __HIP_MEMORY_SCOPE_AGENT) < target) {
          __builtin_amdgcn_s_sleep(2);
        }
      }
    }
    __syncthreads();
    __builtin_amdgcn_fence(__ATOMIC_ACQUIRE, "agent");  // invalidate WGP$ for u reads
  }
}

// ---------------------------------------------------------------------------
// Tiled (t,n)->(n,t) transpose from (t,m,n) staging to (m,n,t+1) output layout.
// gridDim = (16*64 tiles, M, 2{x,u}); 32x32 tile through LDS, coalesced both ways.
// ---------------------------------------------------------------------------
__global__ __launch_bounds__(256) void transpose_kernel(
    const float* __restrict__ xstage, const float* __restrict__ ustage,
    float* __restrict__ out) {
  __shared__ float tile[32][33];
  const int m = blockIdx.y;
  const int t0 = (blockIdx.x & 15) * 32;
  const int n0 = (blockIdx.x >> 4) * 32;
  const float* src = blockIdx.z ? ustage : xstage;
  float* dst = out + (blockIdx.z ? (size_t)M_ * N_ * TP1_ : (size_t)0);
  const int tx = threadIdx.x & 31;
  const int ty = threadIdx.x >> 5;
#pragma unroll
  for (int i = 0; i < 4; ++i) {
    int t = t0 + ty + i * 8;
    tile[ty + i * 8][tx] = src[(size_t)t * (M_ * N_) + (size_t)m * N_ + n0 + tx];
  }
  __syncthreads();
#pragma unroll
  for (int i = 0; i < 4; ++i) {
    int n = n0 + ty + i * 8;
    dst[(size_t)m * N_ * TP1_ + (size_t)n * TP1_ + 1 + t0 + tx] =
        tile[tx][ty + i * 8];
  }
}

// ---------------------------------------------------------------------------
// y head: y[m,o,t] = sum_n u_full[m,n,t+1] * W_out[o,n] + b_out[o]
// One WG per m; lanes parallel over t -> contiguous reads along t.
// ---------------------------------------------------------------------------
__global__ __launch_bounds__(256) void y_kernel(
    const float* __restrict__ u_full, const float* __restrict__ W_out,
    const float* __restrict__ b_out, float* __restrict__ y) {
  const int m = blockIdx.x;
  const int t0 = threadIdx.x;
  float a0 = 0.f, a1 = 0.f, b0 = 0.f, b1 = 0.f;
  const float* up = u_full + (size_t)m * N_ * TP1_;
  for (int n = 0; n < N_; ++n) {
    float w0 = W_out[n];
    float w1 = W_out[N_ + n];
    float ua = up[(size_t)n * TP1_ + 1 + t0];
    float ub = up[(size_t)n * TP1_ + 1 + t0 + 256];
    a0 += w0 * ua; a1 += w1 * ua;
    b0 += w0 * ub; b1 += w1 * ub;
  }
  float* yp = y + (size_t)m * NOUT_ * T_;
  yp[t0] = a0 + b_out[0];
  yp[T_ + t0] = a1 + b_out[1];
  yp[t0 + 256] = b0 + b_out[0];
  yp[T_ + t0 + 256] = b1 + b_out[1];
}

// ---------------------------------------------------------------------------
extern "C" void kernel_launch(void* const* d_in, const int* in_sizes, int n_in,
                              void* d_out, int out_size, void* d_ws, size_t ws_size,
                              hipStream_t stream) {
  (void)in_sizes; (void)n_in; (void)out_size;
  const float* I    = (const float*)d_in[0];
  const float* x0   = (const float*)d_in[1];
  const float* W_in = (const float*)d_in[2];
  const float* Wrec = (const float*)d_in[3];
  const float* brec = (const float*)d_in[4];
  const float* Wout = (const float*)d_in[5];
  const float* bout = (const float*)d_in[6];
  float* out = (float*)d_out;

  char* ws = (char*)d_ws;
  size_t off = 0;
  unsigned* bar = (unsigned*)(ws + off); off = 256;
  unsigned short* ubuf0 = (unsigned short*)(ws + off); off += (size_t)M_ * N_ * 2;
  unsigned short* ubuf1 = (unsigned short*)(ws + off); off += (size_t)M_ * N_ * 2;
  off = (off + 255) & ~(size_t)255;
  unsigned short* wrecbf = (unsigned short*)(ws + off); off += (size_t)N_ * N_ * 2;
  off = (off + 255) & ~(size_t)255;
  const size_t stage_elems = (size_t)T_ * M_ * N_;
  float* xstage = (float*)(ws + off);
  float* ustage = xstage + stage_elems;
  const bool staged = ws_size >= off + 2 * stage_elems * sizeof(float);

  const size_t xu_plane = (size_t)M_ * N_ * TP1_;

  // 1) init / convert / barrier reset
  {
    int nblk = (int)(((size_t)N_ * N_ + 255) / 256);
    hipLaunchKernelGGL(init_kernel, dim3(nblk), dim3(256), 0, stream,
                       Wrec, x0, wrecbf, ubuf0, out, bar);
  }

  // 2) persistent RNN loop
  float* xo; float* uo; unsigned long long mstr, nstr, tstr;
  if (staged) {
    xo = xstage; uo = ustage;
    mstr = N_; nstr = 1; tstr = (unsigned long long)M_ * N_;
  } else {
    xo = out + 1; uo = out + xu_plane + 1;         // (+1): t+1 base
    mstr = (unsigned long long)N_ * TP1_; nstr = TP1_; tstr = 1;
  }
  hipLaunchKernelGGL(rnn_step_kernel, dim3(NWG_), dim3(256), 0, stream,
                     I, x0, W_in, brec, wrecbf, ubuf0, ubuf1,
                     xo, uo, mstr, nstr, tstr, bar);

  // 3) layout transpose (only when staged in ws)
  if (staged) {
    hipLaunchKernelGGL(transpose_kernel, dim3(1024, M_, 2), dim3(256), 0, stream,
                       xstage, ustage, out);
  }

  // 4) output head
  hipLaunchKernelGGL(y_kernel, dim3(M_), dim3(256), 0, stream,
                     out + xu_plane, Wout, bout, out + 2 * xu_plane);
}
